// GraphAttentionLayer_57269093925479
// MI455X (gfx1250) — compile-verified, matrix operand order checked by reading, not run on hardware
//
#include <hip/hip_runtime.h>
#include <hip/hip_bf16.h>

#define NB    8
#define NN    2048
#define FD    256
#define ALPHA 0.2f
#define NEGV  -9.0e15f
#define EPSV  1e-5f

typedef __attribute__((ext_vector_type(2)))  float    v2f;
typedef __attribute__((ext_vector_type(8)))  float    v8f;
typedef __attribute__((ext_vector_type(4)))  int      v4i;
typedef __attribute__((ext_vector_type(16))) _Float16 v16h;

// ---- lane-XOR butterfly via DS_SWIZZLE (single ds op, no VALU mask setup) ----
#if __has_builtin(__builtin_amdgcn_ds_swizzle)
#define XORF(v, m) __int_as_float(__builtin_amdgcn_ds_swizzle(__float_as_int(v), (0x1f | ((m) << 10))))
#else
#define XORF(v, m) __shfl_xor((v), (m), 32)
#endif

// ---- async global->LDS (ASYNCcnt); load builtin proven present on this toolchain ----
#if __has_builtin(__builtin_amdgcn_global_load_async_to_lds_b128)
#define HAVE_ASYNC 1
#else
#define HAVE_ASYNC 0
#endif

#define AS1 __attribute__((address_space(1)))
#define AS3 __attribute__((address_space(3)))

__device__ __forceinline__ void issue_adj16(const float* gsrc, float* lslot) {
#if HAVE_ASYNC
  __builtin_amdgcn_global_load_async_to_lds_b128((AS1 v4i*)gsrc, (AS3 v4i*)lslot, 0, 0);
#else
  *(float4*)lslot = *(const float4*)gsrc;
#endif
}

__device__ __forceinline__ void wait_adj() {
#if HAVE_ASYNC
#if __has_builtin(__builtin_amdgcn_s_wait_asynccnt)
  __builtin_amdgcn_s_wait_asynccnt(3);   // DEPTH-1: oldest chunk has landed
#else
  asm volatile("s_wait_asynccnt 0x3" ::: "memory");
#endif
#endif
}

// ---------------------------------------------------------------- k_zero
__global__ __launch_bounds__(256) void k_zero(float* p, int n) {
  int i = blockIdx.x * 256 + threadIdx.x;
  if (i < n) p[i] = 0.f;
}

// ---------------------------------------------------------------- k_proj
// Wh = h @ W  (fp32 WMMA 16x16x4), writes WhT[b][f][n] in f16,
// atomically accumulates s1[b][n] = Wh.a1, s2[b][n] = Wh.a2
#define HSTR 258   // padded fp32 row stride for 16x256 h tile

__global__ __launch_bounds__(256) void k_proj(const float* __restrict__ h,
                                              const float* __restrict__ W,
                                              const float* __restrict__ a,
                                              _Float16* __restrict__ whT,
                                              float* __restrict__ s1,
                                              float* __restrict__ s2) {
  __shared__ float hs[16 * HSTR];
  const int b   = blockIdx.x;          // 0..7
  const int n0  = blockIdx.y * 16;     // row tile
  const int tid = threadIdx.x;

  // cooperative load of 16x256 h tile (4096 floats, float4 per thread x4)
  const float* hbase = h + ((size_t)b * NN + n0) * FD;
  #pragma unroll
  for (int i = 0; i < 4; ++i) {
    int idx  = tid + i * 256;          // float4 index 0..1023
    int row  = idx >> 6;               // 64 float4 per row
    int col4 = idx & 63;
    float4 v = *(const float4*)(hbase + row * FD + col4 * 4);
    float* dst = &hs[row * HSTR + col4 * 4];
    dst[0] = v.x; dst[1] = v.y; dst[2] = v.z; dst[3] = v.w;
  }
  __syncthreads();

  const int wave = tid >> 5;
  const int lane = tid & 31;
  const int mrow = lane & 15;
  const int hi   = lane >> 4;          // 0 or 1
  const int khi  = hi * 2;             // K sub-offset for f32 A/B fragments

  #pragma unroll
  for (int t = 0; t < 2; ++t) {
    const int c0  = (wave + t * 8) * 16;
    const int col = c0 + mrow;
    v8f acc = {};
#if __has_builtin(__builtin_amdgcn_wmma_f32_16x16x4_f32)
    for (int k = 0; k < FD; k += 4) {
      v2f av, bv;
      av.x = hs[mrow * HSTR + k + khi + 0];
      av.y = hs[mrow * HSTR + k + khi + 1];
      bv.x = W[(k + khi + 0) * FD + col];
      bv.y = W[(k + khi + 1) * FD + col];
      acc = __builtin_amdgcn_wmma_f32_16x16x4_f32(false, av, false, bv,
                                                  (short)0, acc, false, false);
    }
#else
    #pragma unroll
    for (int r = 0; r < 8; ++r) {
      int m = r + 8 * hi;
      float s = 0.f;
      for (int k = 0; k < FD; ++k) s += hs[m * HSTR + k] * W[k * FD + col];
      acc[r] = s;
    }
#endif
    const float a1c = a[col];
    const float a2c = a[FD + col];
    #pragma unroll
    for (int r = 0; r < 8; ++r) {
      const int m = r + 8 * hi;
      const float x = acc[r];
      whT[((size_t)b * FD + col) * NN + n0 + m] = (_Float16)x;
      float p1 = x * a1c, p2 = x * a2c;
      p1 += XORF(p1, 8); p2 += XORF(p2, 8);
      p1 += XORF(p1, 4); p2 += XORF(p2, 4);
      p1 += XORF(p1, 2); p2 += XORF(p2, 2);
      p1 += XORF(p1, 1); p2 += XORF(p2, 1);
      if ((lane & 15) == 0) {
        atomicAdd(&s1[b * NN + n0 + m], p1);
        atomicAdd(&s2[b * NN + n0 + m], p2);
      }
    }
  }
}

// ---------------------------------------------------------------- k_attn
// Flash-style: out = gelu(layernorm(softmax(mask(leaky(s1+s2'))) @ Wh))
#define ROWS  32
#define CK    32
#define PSTR  40     // f16 halfs per padded P row (80B, 16B aligned)
#define DEPTH 4      // async adj ring depth (4 x 4KB LDS)

__global__ __launch_bounds__(256) void k_attn(const float* __restrict__ adj,
                                              const _Float16* __restrict__ whT,
                                              const float* __restrict__ s1,
                                              const float* __restrict__ s2,
                                              const float* __restrict__ gamma,
                                              const float* __restrict__ beta,
                                              float* __restrict__ out) {
  __shared__ alignas(16) float    adjbuf[DEPTH * 256 * 4];   // per-thread float4 slots
  __shared__ alignas(16) _Float16 P[ROWS * PSTR];
  __shared__ float mrow[ROWS], lrow[ROWS], sscale[ROWS];
  __shared__ float rsum[ROWS], rsumsq[ROWS];

  const int b   = blockIdx.x;
  const int i0  = blockIdx.y * ROWS;
  const int tid = threadIdx.x;
  const int lane = tid & 31;
  const int wave = tid >> 5;

  if (tid < ROWS) {
    mrow[tid] = -3.0e38f; lrow[tid] = 0.f;
    rsum[tid] = 0.f;      rsumsq[tid] = 0.f;
  }

  // wave tiling: 2 M-tiles x 4 N-quads; each wave: 16 rows x 64 cols
  const int mbase = (wave >> 2) * 16;
  const int f0    = (wave & 3) * 64;
  const int hi    = lane >> 4;
  const int kb    = hi * 8;                 // A/B K sub-offset (halfs)
  const int arow  = mbase + (lane & 15);
  v8f acc[4] = {};

  // score-phase mapping: 8 threads per row, 4 cols each
  const int   qrow = tid >> 3;
  const int   qcol = (tid & 7) * 4;
  const float s1v  = s1[b * NN + i0 + qrow];
  const float* adjrow = adj + ((size_t)b * NN + i0 + qrow) * NN;
  const float* s2b    = s2 + b * NN;
  float* myslot0 = &adjbuf[tid * 4];

  // prime the async ring (each thread owns its 16B slot -> per-wave wait is enough)
  #pragma unroll
  for (int d = 0; d < DEPTH; ++d)
    issue_adj16(adjrow + d * CK + qcol, myslot0 + d * 1024);
  __syncthreads();

  int it = 0;
  for (int j0 = 0; j0 < NN; j0 += CK, ++it) {
    const int slot = it & (DEPTH - 1);
    // ---- phase A: masked leaky scores + online softmax + P -> LDS ----
    wait_adj();
    float4 am  = *(const float4*)(myslot0 + slot * 1024);
    float4 sv  = *(const float4*)(s2b + j0 + qcol);
    float e0 = s1v + sv.x; e0 = e0 > 0.f ? e0 : ALPHA * e0; e0 = am.x > 0.f ? e0 : NEGV;
    float e1 = s1v + sv.y; e1 = e1 > 0.f ? e1 : ALPHA * e1; e1 = am.y > 0.f ? e1 : NEGV;
    float e2 = s1v + sv.z; e2 = e2 > 0.f ? e2 : ALPHA * e2; e2 = am.z > 0.f ? e2 : NEGV;
    float e3 = s1v + sv.w; e3 = e3 > 0.f ? e3 : ALPHA * e3; e3 = am.w > 0.f ? e3 : NEGV;

    // refill the just-consumed slot (wrapped addresses on the tail; never read)
    {
      int jn = j0 + DEPTH * CK;
      if (jn >= NN) jn -= NN;
      issue_adj16(adjrow + jn + qcol, myslot0 + slot * 1024);
    }

    float cmax = fmaxf(fmaxf(e0, e1), fmaxf(e2, e3));
    cmax = fmaxf(cmax, XORF(cmax, 4));
    cmax = fmaxf(cmax, XORF(cmax, 2));
    cmax = fmaxf(cmax, XORF(cmax, 1));
    const float mold = mrow[qrow];
    const float mnew = fmaxf(mold, cmax);
    float p0 = __expf(e0 - mnew), p1 = __expf(e1 - mnew);
    float p2 = __expf(e2 - mnew), p3 = __expf(e3 - mnew);
    float psum = (p0 + p1) + (p2 + p3);
    psum += XORF(psum, 4);
    psum += XORF(psum, 2);
    psum += XORF(psum, 1);
    if ((tid & 7) == 0) {
      const float sc = __expf(mold - mnew);
      sscale[qrow] = sc;
      mrow[qrow]   = mnew;
      lrow[qrow]   = lrow[qrow] * sc + psum;
    }
    union { _Float16 hh[4]; uint2 u; } up;
    up.hh[0] = (_Float16)p0; up.hh[1] = (_Float16)p1;
    up.hh[2] = (_Float16)p2; up.hh[3] = (_Float16)p3;
    *(uint2*)&P[qrow * PSTR + qcol] = up.u;
    __syncthreads();

    // ---- phase B: rescale accumulators, A from LDS, B from WhT, WMMA ----
    #pragma unroll
    for (int r = 0; r < 8; ++r) {
      const float sc = sscale[mbase + r + 8 * hi];
      acc[0][r] *= sc; acc[1][r] *= sc; acc[2][r] *= sc; acc[3][r] *= sc;
    }
    union { uint4 q[2]; v16h h; } ua;
    ua.q[0] = *(const uint4*)&P[arow * PSTR + kb];
    ua.q[1] = *(const uint4*)&P[arow * PSTR + kb + 16];
    #pragma unroll
    for (int t = 0; t < 4; ++t) {
      const int col = f0 + t * 16 + (lane & 15);
      const _Float16* src = whT + ((size_t)b * FD + col) * NN + j0 + kb;
      union { uint4 q[2]; v16h h; } ub;
      ub.q[0] = *(const uint4*)(src);
      ub.q[1] = *(const uint4*)(src + 16);
      acc[t] = __builtin_amdgcn_wmma_f32_16x16x32_f16(false, ua.h, false, ub.h,
                                                      (short)0, acc[t], false, false);
    }
    __syncthreads();
  }

  // ---- finalize: /l, layernorm stats across the 4 waves of each M-tile ----
  #pragma unroll
  for (int r = 0; r < 8; ++r) {
    const int m = mbase + r + 8 * hi;
    const float linv = 1.f / lrow[m];
    float ps = 0.f, pq = 0.f;
    #pragma unroll
    for (int t = 0; t < 4; ++t) {
      const float x = acc[t][r] * linv;
      acc[t][r] = x;
      ps += x; pq += x * x;
    }
    ps += XORF(ps, 1); pq += XORF(pq, 1);
    ps += XORF(ps, 2); pq += XORF(pq, 2);
    ps += XORF(ps, 4); pq += XORF(pq, 4);
    ps += XORF(ps, 8); pq += XORF(pq, 8);
    if ((lane & 15) == 0) {
      atomicAdd(&rsum[m], ps);
      atomicAdd(&rsumsq[m], pq);
    }
  }
  __syncthreads();
  #pragma unroll
  for (int r = 0; r < 8; ++r) {
    const int m   = mbase + r + 8 * hi;
    const float mu  = rsum[m] * (1.f / FD);
    const float var = rsumsq[m] * (1.f / FD) - mu * mu;
    const float rst = rsqrtf(var + EPSV);
    #pragma unroll
    for (int t = 0; t < 4; ++t) {
      const int col = f0 + t * 16 + (lane & 15);
      const float y = (acc[t][r] - mu) * rst * gamma[col] + beta[col];
      const float g = 0.5f * y * (1.f + erff(y * 0.70710678118f));
      out[((size_t)b * NN + i0 + m) * FD + col] = g;
    }
  }
}

// ---------------------------------------------------------------- launch
extern "C" void kernel_launch(void* const* d_in, const int* in_sizes, int n_in,
                              void* d_out, int out_size, void* d_ws, size_t ws_size,
                              hipStream_t stream) {
  const float* h     = (const float*)d_in[0];
  const float* adj   = (const float*)d_in[1];
  const float* W     = (const float*)d_in[2];
  const float* a     = (const float*)d_in[3];
  const float* gamma = (const float*)d_in[4];
  const float* beta  = (const float*)d_in[5];
  float* out = (float*)d_out;

  const size_t whtBytes = (size_t)NB * FD * NN * sizeof(_Float16); // 8 MB
  _Float16* whT = (_Float16*)d_ws;
  float* s1 = (float*)((char*)d_ws + whtBytes);
  float* s2 = s1 + (size_t)NB * NN;

  k_zero<<<(2 * NB * NN + 255) / 256, 256, 0, stream>>>(s1, 2 * NB * NN);
  k_proj<<<dim3(NB, NN / 16), 256, 0, stream>>>(h, W, a, whT, s1, s2);
  k_attn<<<dim3(NB, NN / ROWS), 256, 0, stream>>>(adj, whT, s1, s2, gamma, beta, out);
}
